// TransformerBlock_75771813037017
// MI455X (gfx1250) — compile-verified, matrix operand order checked by reading, not run on hardware
//
#include <hip/hip_runtime.h>
#include <hip/hip_bf16.h>

typedef __attribute__((ext_vector_type(8)))  _Float16 v8h;
typedef __attribute__((ext_vector_type(16))) _Float16 v16h;
typedef __attribute__((ext_vector_type(8)))  float    v8f;

#define EMB 1024
#define FFD 1024
#define BSZ 4
#define SEQ 2048

// ---------------------------------------------------------------------------
// Generic WMMA f16 GEMM:  C[b] = A[b] (MxK, row-major) * Bt[b]^T (Bt is NxK,
// row-major) with optional scale/bias/residual/relu and f32 / f16 /
// transposed-f16 stores. Block = 256 threads (8 waves); each wave computes a
// 32x64 output tile (2x4 WMMA subtiles); block tile = 128x128.
// ---------------------------------------------------------------------------
__global__ __launch_bounds__(256)
void gemm_wmma_f16_tb75771813037017(
    const _Float16* __restrict__ A, int lda, long sA,
    const _Float16* __restrict__ Bt, int ldb, long sB,
    const float* __restrict__ bias,
    float*    __restrict__ Cf,
    _Float16* __restrict__ Ch,
    _Float16* __restrict__ ChT,
    int ldc, long sC, int ldcT, long sCT, int transS,
    int M, int N, int K,
    float scale, int relu,
    const float* __restrict__ R)
{
    const int lane  = threadIdx.x & 31;
    const int wave  = threadIdx.x >> 5;
    const int wm    = wave & 3;            // 4 waves along M  -> 128 rows
    const int wn    = wave >> 2;           // 2 waves along N  -> 128 cols
    const int tileM = blockIdx.y * 128 + wm * 32;
    const int tileN = blockIdx.x * 128 + wn * 64;
    const long b    = blockIdx.z;

    const _Float16* Ab = A  + b * sA;
    const _Float16* Bb = Bt + b * sB;

    const int half_ = lane >> 4;   // lane-half selects K sub-range per ISA layout
    const int l16   = lane & 15;

    v8f acc[2][4];
#pragma unroll
    for (int mi = 0; mi < 2; ++mi)
#pragma unroll
        for (int ni = 0; ni < 4; ++ni)
#pragma unroll
            for (int i = 0; i < 8; ++i) acc[mi][ni][i] = 0.0f;

    for (int k0 = 0; k0 < K; k0 += 32) {
        // A fragments: per lane, rows tileM+mi*16+l16, K chunks [half*8, half*8+8)
        // and [16+half*8, 16+half*8+8)  (ISA 16-bit A 16x32 layout)
        v16h afrag[2];
#pragma unroll
        for (int mi = 0; mi < 2; ++mi) {
            const _Float16* ap = Ab + (long)(tileM + mi * 16 + l16) * lda + k0 + half_ * 8;
            v8h lo = *(const v8h*)ap;
            v8h hi = *(const v8h*)(ap + 16);
            v16h f;
#pragma unroll
            for (int i = 0; i < 8; ++i) { f[i] = lo[i]; f[i + 8] = hi[i]; }
            afrag[mi] = f;
        }
#pragma unroll
        for (int ni = 0; ni < 4; ++ni) {
            const _Float16* bp = Bb + (long)(tileN + ni * 16 + l16) * ldb + k0 + half_ * 8;
            v8h lo = *(const v8h*)bp;
            v8h hi = *(const v8h*)(bp + 16);
            v16h f;
#pragma unroll
            for (int i = 0; i < 8; ++i) { f[i] = lo[i]; f[i + 8] = hi[i]; }
#pragma unroll
            for (int mi = 0; mi < 2; ++mi)
                acc[mi][ni] = __builtin_amdgcn_wmma_f32_16x16x32_f16(
                    false, afrag[mi], false, f, (short)0, acc[mi][ni], false, false);
        }
    }

    // Epilogue. C/D layout: VGPR r, lane L -> M = r + 8*(L/16), N = L%16.
#pragma unroll
    for (int mi = 0; mi < 2; ++mi)
#pragma unroll
        for (int ni = 0; ni < 4; ++ni)
#pragma unroll
            for (int r = 0; r < 8; ++r) {
                const int gm = tileM + mi * 16 + r + 8 * half_;
                const int gn = tileN + ni * 16 + l16;
                float v = acc[mi][ni][r] * scale;
                if (bias) v += bias[gn];
                if (R)    v += R[b * sC + (long)gm * ldc + gn];
                if (relu) v = v > 0.0f ? v : 0.0f;
                if (Cf) Cf[b * sC + (long)gm * ldc + gn] = v;
                if (Ch) Ch[b * sC + (long)gm * ldc + gn] = (_Float16)v;
                if (ChT) {
                    const long bb = gm / transS;
                    ChT[bb * sCT + (long)gn * ldcT + (gm % transS)] = (_Float16)v;
                }
            }
}

// ---------------------------------------------------------------------------
// LayerNorm (row length C==1024) -> f16 output. One block per row.
// ---------------------------------------------------------------------------
__global__ __launch_bounds__(256)
void ln_f16_tb75771813037017(const float* __restrict__ x,
                             const float* __restrict__ g,
                             const float* __restrict__ be,
                             _Float16* __restrict__ out, int C)
{
    __shared__ float r1[8], r2[8];
    const long row = blockIdx.x;
    const float* xr = x + row * (long)C;
    const int tid = threadIdx.x, lane = tid & 31, wid = tid >> 5;

    float s = 0.0f, ss = 0.0f;
    for (int i = tid; i < C; i += 256) { float v = xr[i]; s += v; ss += v * v; }
    for (int off = 16; off; off >>= 1) {
        s  += __shfl_xor(s,  off, 32);
        ss += __shfl_xor(ss, off, 32);
    }
    if (lane == 0) { r1[wid] = s; r2[wid] = ss; }
    __syncthreads();
    s = 0.0f; ss = 0.0f;
#pragma unroll
    for (int w = 0; w < 8; ++w) { s += r1[w]; ss += r2[w]; }
    const float mu   = s / C;
    const float var  = ss / C - mu * mu;
    const float rstd = rsqrtf(var + 1e-5f);
    _Float16* orow = out + row * (long)C;
    for (int i = tid; i < C; i += 256)
        orow[i] = (_Float16)((xr[i] - mu) * rstd * g[i] + be[i]);
}

// ---------------------------------------------------------------------------
// Weight convert + transpose: W [K][N] f32 -> WT [N][K] f16. 32x32 LDS tiles.
// ---------------------------------------------------------------------------
__global__ __launch_bounds__(256)
void transpose_w_tb75771813037017(const float* __restrict__ W,
                                  _Float16* __restrict__ WT, int K, int N)
{
    __shared__ _Float16 tile[32][33];
    const int bk = blockIdx.y * 32, bn = blockIdx.x * 32;
    const int tx = threadIdx.x & 31, ty = threadIdx.x >> 5;   // 32x8
#pragma unroll
    for (int i = 0; i < 32; i += 8)
        tile[ty + i][tx] = (_Float16)W[(long)(bk + ty + i) * N + bn + tx];
    __syncthreads();
#pragma unroll
    for (int i = 0; i < 32; i += 8)
        WT[(long)(bn + ty + i) * K + bk + tx] = tile[tx][ty + i];
}

// ---------------------------------------------------------------------------
// Row softmax (row length T==2048), in-place f32 + f16 copy. One block/row.
// ---------------------------------------------------------------------------
__global__ __launch_bounds__(256)
void softmax_tb75771813037017(float* __restrict__ S,
                              _Float16* __restrict__ Wh, int T)
{
    __shared__ float red[8];
    const long row = blockIdx.x;
    float* sr = S + row * (long)T;
    const int tid = threadIdx.x, lane = tid & 31, wid = tid >> 5;

    float m = -3.0e38f;
    for (int i = tid; i < T; i += 256) m = fmaxf(m, sr[i]);
    for (int off = 16; off; off >>= 1) m = fmaxf(m, __shfl_xor(m, off, 32));
    if (lane == 0) red[wid] = m;
    __syncthreads();
    m = red[0];
#pragma unroll
    for (int w = 1; w < 8; ++w) m = fmaxf(m, red[w]);
    __syncthreads();

    float sum = 0.0f;
    for (int i = tid; i < T; i += 256) {
        float e = __expf(sr[i] - m);
        sr[i] = e;
        sum += e;
    }
    for (int off = 16; off; off >>= 1) sum += __shfl_xor(sum, off, 32);
    if (lane == 0) red[wid] = sum;
    __syncthreads();
    sum = 0.0f;
#pragma unroll
    for (int w = 0; w < 8; ++w) sum += red[w];
    const float inv = 1.0f / sum;
    _Float16* wr = Wh + row * (long)T;
    for (int i = tid; i < T; i += 256) {
        float v = sr[i] * inv;
        sr[i] = v;
        wr[i] = (_Float16)v;
    }
}

// ---------------------------------------------------------------------------
extern "C" void kernel_launch(void* const* d_in, const int* in_sizes, int n_in,
                              void* d_out, int out_size, void* d_ws, size_t ws_size,
                              hipStream_t stream) {
    const float* x   = (const float*)d_in[0];
    const float* Wq  = (const float*)d_in[1];
    const float* bq  = (const float*)d_in[2];
    const float* Wk  = (const float*)d_in[3];
    const float* bk  = (const float*)d_in[4];
    const float* Wv  = (const float*)d_in[5];
    const float* bv  = (const float*)d_in[6];
    const float* Wo  = (const float*)d_in[7];
    const float* bo  = (const float*)d_in[8];
    const float* W1  = (const float*)d_in[9];
    const float* b1  = (const float*)d_in[10];
    const float* W2  = (const float*)d_in[11];
    const float* b2  = (const float*)d_in[12];
    const float* g1  = (const float*)d_in[13];
    const float* be1 = (const float*)d_in[14];
    const float* g2  = (const float*)d_in[15];
    const float* be2 = (const float*)d_in[16];

    const int  E  = EMB, S = SEQ, B = BSZ, F = FFD;
    const long BS = (long)B * S;                     // 8192 rows

    char* ws = (char*)d_ws;
    const size_t MB = 1024ull * 1024ull;
    // workspace layout (aliased; peak ~108 MB)
    _Float16* wqT  = (_Float16*)(ws +  0 * MB);      // 2 MB each
    _Float16* wkT  = (_Float16*)(ws +  2 * MB);
    _Float16* wvT  = (_Float16*)(ws +  4 * MB);
    _Float16* woT  = (_Float16*)(ws +  6 * MB);
    _Float16* w1T  = (_Float16*)(ws +  8 * MB);
    _Float16* w2T  = (_Float16*)(ws + 10 * MB);
    _Float16* xn   = (_Float16*)(ws + 12 * MB);      // 16 MB
    _Float16* qh   = (_Float16*)(ws + 28 * MB);      // 16 MB
    _Float16* kh   = (_Float16*)(ws + 44 * MB);      // 16 MB
    _Float16* vT   = (_Float16*)(ws + 60 * MB);      // 16 MB  [B][E][S]
    float*    x2   = (float*)   (ws + 76 * MB);      // 32 MB
    _Float16* wath = (_Float16*)(ws + 28 * MB);      // 32 MB (reuses q+k)
    _Float16* attn = (_Float16*)(ws + 12 * MB);      // 16 MB (reuses xn)
    _Float16* xn2  = (_Float16*)(ws + 28 * MB);      // 16 MB (reuses q)
    _Float16* hh   = (_Float16*)(ws + 44 * MB);      // 16 MB (reuses k)

    float* outp = (float*)d_out;
    float* wout = outp + BS * E;                     // softmax weights region

    dim3 blk(256);

    // 1) weights -> f16 transposed [N][K]
    {
        dim3 g(E / 32, E / 32);
        transpose_w_tb75771813037017<<<g, blk, 0, stream>>>(Wq, wqT, E, E);
        transpose_w_tb75771813037017<<<g, blk, 0, stream>>>(Wk, wkT, E, E);
        transpose_w_tb75771813037017<<<g, blk, 0, stream>>>(Wv, wvT, E, E);
        transpose_w_tb75771813037017<<<g, blk, 0, stream>>>(Wo, woT, E, E);
        transpose_w_tb75771813037017<<<dim3(F / 32, E / 32), blk, 0, stream>>>(W1, w1T, E, F);
        transpose_w_tb75771813037017<<<dim3(E / 32, F / 32), blk, 0, stream>>>(W2, w2T, F, E);
    }

    // 2) xn = LN(x) -> f16
    ln_f16_tb75771813037017<<<dim3((unsigned)BS), blk, 0, stream>>>(x, g1, be1, xn, E);

    // 3) q, k projections ; v stored transposed [B][E][S]
    dim3 gp(E / 128, (unsigned)(BS / 128), 1);
    gemm_wmma_f16_tb75771813037017<<<gp, blk, 0, stream>>>(
        xn, E, 0, wqT, E, 0, bq, nullptr, qh, nullptr,
        E, 0, 0, 0, 1, (int)BS, E, E, 1.0f, 0, nullptr);
    gemm_wmma_f16_tb75771813037017<<<gp, blk, 0, stream>>>(
        xn, E, 0, wkT, E, 0, bk, nullptr, kh, nullptr,
        E, 0, 0, 0, 1, (int)BS, E, E, 1.0f, 0, nullptr);
    gemm_wmma_f16_tb75771813037017<<<gp, blk, 0, stream>>>(
        xn, E, 0, wvT, E, 0, bv, nullptr, nullptr, vT,
        E, 0, /*ldcT=*/S, /*sCT=*/(long)E * S, /*transS=*/S,
        (int)BS, E, E, 1.0f, 0, nullptr);

    // 4) scores = q @ k^T / 32  -> f32 directly into d_out w-region
    dim3 gs(S / 128, S / 128, B);
    gemm_wmma_f16_tb75771813037017<<<gs, blk, 0, stream>>>(
        qh, E, (long)S * E, kh, E, (long)S * E, nullptr,
        wout, nullptr, nullptr, S, (long)S * S, 0, 0, 1,
        S, S, E, 0.03125f, 0, nullptr);

    // 5) softmax in-place (f32) + f16 copy
    softmax_tb75771813037017<<<dim3((unsigned)BS), blk, 0, stream>>>(wout, wath, S);

    // 6) attn = w @ v   (Bt = vT [B][E][S])
    dim3 ga(E / 128, S / 128, B);
    gemm_wmma_f16_tb75771813037017<<<ga, blk, 0, stream>>>(
        wath, S, (long)S * S, vT, S, (long)E * S, nullptr,
        nullptr, attn, nullptr, E, (long)S * E, 0, 0, 1,
        S, E, S, 1.0f, 0, nullptr);

    // 7) x2 = x + attn @ Wo + bo   (f32)
    gemm_wmma_f16_tb75771813037017<<<gp, blk, 0, stream>>>(
        attn, E, 0, woT, E, 0, bo, x2, nullptr, nullptr,
        E, 0, 0, 0, 1, (int)BS, E, E, 1.0f, 0, x);

    // 8) xn2 = LN(x2) -> f16
    ln_f16_tb75771813037017<<<dim3((unsigned)BS), blk, 0, stream>>>(x2, g2, be2, xn2, E);

    // 9) h = relu(xn2 @ W1 + b1) -> f16
    gemm_wmma_f16_tb75771813037017<<<dim3(F / 128, (unsigned)(BS / 128), 1), blk, 0, stream>>>(
        xn2, E, 0, w1T, E, 0, b1, nullptr, hh, nullptr,
        F, 0, 0, 0, 1, (int)BS, F, E, 1.0f, 1, nullptr);

    // 10) out = x2 + h @ W2 + b2  (f32, to d_out)
    gemm_wmma_f16_tb75771813037017<<<gp, blk, 0, stream>>>(
        hh, F, 0, w2T, F, 0, b2, outp, nullptr, nullptr,
        E, 0, 0, 0, 1, (int)BS, E, F, 1.0f, 0, x2);
}
